// Encoder_61649960567159
// MI455X (gfx1250) — matrix operand order repaired and checked by reading, not compiled, not run on hardware
//
#include <hip/hip_runtime.h>
#include <hip/hip_bf16.h>

typedef __attribute__((ext_vector_type(16))) _Float16 v16h;
typedef __attribute__((ext_vector_type(8)))  float    v8f;

// ---------------------------------------------------------------------------
// Utility: fill a float buffer with a constant (d_out / d_ws are poisoned).
// ---------------------------------------------------------------------------
__global__ void fill_kernel(float* __restrict__ p, float v, long long n) {
    long long i = (long long)blockIdx.x * blockDim.x + threadIdx.x;
    if (i < n) p[i] = v;
}

// ---------------------------------------------------------------------------
// deg[dst[e]] += edge_attr[e]   (deg pre-initialized to 1.0 = self-loop w)
// ---------------------------------------------------------------------------
__global__ void degree_kernel(const int* __restrict__ dst,
                              const float* __restrict__ ea,
                              float* __restrict__ deg, int E) {
    int e = blockIdx.x * blockDim.x + threadIdx.x;
    if (e < E) atomicAdd(&deg[dst[e]], ea[e]);
}

// deg -> dis = deg>0 ? rsqrt(max(deg,1e-12)) : 0   (in place)
__global__ void invsqrt_kernel(float* __restrict__ deg, int N) {
    int i = blockIdx.x * blockDim.x + threadIdx.x;
    if (i < N) {
        float d = deg[i];
        deg[i] = d > 0.f ? rsqrtf(fmaxf(d, 1e-12f)) : 0.f;
    }
}

// ---------------------------------------------------------------------------
// Wt[n][k] = (f16) W[k][n]  — one-time transpose+convert of a weight matrix
// so GEMM B-fragments are contiguous 32B v16h loads.
// ---------------------------------------------------------------------------
__global__ void transpose_f16_kernel(const float* __restrict__ W,
                                     _Float16* __restrict__ Wt,
                                     int K, int Nn) {
    int idx = blockIdx.x * blockDim.x + threadIdx.x;
    if (idx >= K * Nn) return;
    int n = idx / K;
    int k = idx - n * K;
    Wt[idx] = (_Float16)W[(size_t)k * Nn + n];
}

// ---------------------------------------------------------------------------
// C[M x Nn] = A[M x K] @ B[K x Nn]  via v_wmma_f32_16x16x32_f16.
// B is supplied pre-transposed + converted: Bt[Nn x K] in f16.
// One wave computes a 16x64 strip (4 N-tiles), reusing its A fragment 4x.
// All 4 B fragments are loaded as one clause BEFORE the 4 WMMAs so the
// s_wait_loadcnt values are partial and XDL work overlaps outstanding loads.
// Requires M%16==0, Nn%64==0, K%32==0 (M=50000, K in {512,256}, Nn in {256,128}).
//
// VGPR layouts used by this kernel (wave32; contiguous k, same order in A and B):
//   A 16x32 f16 : lane L -> row=L&15, kh=L>>4;
//                 a[j]   = A[row][k0 + 16*kh + j]       (j<8)
//                 a[j+8] = A[row][k0 + 16*kh + 8 + j]   (j<8)
//   B 32x16 f16 : lane L -> col=L&15, kh=L>>4;
//                 b[j]   = B[k0 + 16*kh + j][col] = Bt[col][k0+16*kh+j], j=0..15
//   D 16x16 f32 : lane L, vgpr v -> D[v + 8*kh][L&15]
// ---------------------------------------------------------------------------
template <typename TA>
__global__ void __launch_bounds__(256)
gemm_wmma_f16(const TA* __restrict__ A, const _Float16* __restrict__ Bt,
              float* __restrict__ C, int M, int K, int Nn) {
    const int lane    = threadIdx.x & 31;
    const int wave    = threadIdx.x >> 5;
    const int ngroups = Nn >> 6;                       // groups of 4 N-tiles
    const int gid     = blockIdx.x * (blockDim.x >> 5) + wave;
    const int total   = (M >> 4) * ngroups;
    if (gid >= total) return;                          // wave-uniform guard

    const int tm  = gid / ngroups;
    const int tg  = gid - tm * ngroups;
    const int r   = lane & 15;
    const int kh  = lane >> 4;
    const int row = tm * 16 + r;

    // per-tile B base pointers (column-major f16 panel rows)
    const _Float16* Bp[4];
#pragma unroll
    for (int t = 0; t < 4; ++t)
        Bp[t] = Bt + (size_t)((tg * 4 + t) * 16 + r) * K + kh * 16;

    v8f acc[4] = {};
#pragma unroll 2
    for (int k0 = 0; k0 < K; k0 += 32) {
        // A fragment (converted to f16; no-op when TA == _Float16)
        const TA* Ap = A + (size_t)row * K + k0 + kh * 16;
        v16h a;
#pragma unroll
        for (int j = 0; j < 8; ++j) {
            a[j]     = (_Float16)Ap[j];
            a[j + 8] = (_Float16)Ap[8 + j];
        }
        // issue all 4 B-fragment loads first (one clause) ...
        v16h b[4];
#pragma unroll
        for (int t = 0; t < 4; ++t)
            b[t] = *(const v16h*)(Bp[t] + k0);
        // ... then the 4 WMMAs (partial loadcnt waits, overlap with loads)
#pragma unroll
        for (int t = 0; t < 4; ++t)
            acc[t] = __builtin_amdgcn_wmma_f32_16x16x32_f16(
                false, a, false, b[t], (short)0, acc[t], false, false);
    }

#pragma unroll
    for (int t = 0; t < 4; ++t) {
        const int col = (tg * 4 + t) * 16 + r;
        float* Cp = C + (size_t)(tm * 16 + kh * 8) * Nn + col;
#pragma unroll
        for (int v = 0; v < 8; ++v)
            Cp[(size_t)v * Nn] = acc[t][v];
    }
}

// ---------------------------------------------------------------------------
// Edge scatter, hidden layer:  H[dst] += M1[src] * (dis[src]*w*dis[dst])
// One thread per (edge, 4-column chunk): float4 gather + 4 f32 atomics.
// ---------------------------------------------------------------------------
__global__ void scatter_hid(const float* __restrict__ M1,
                            const int* __restrict__ src,
                            const int* __restrict__ dst,
                            const float* __restrict__ ea,
                            const float* __restrict__ dis,
                            float* __restrict__ H, int E, int C) {
    const int per = C >> 2;
    long long idx = (long long)blockIdx.x * blockDim.x + threadIdx.x;
    if (idx >= (long long)E * per) return;
    int e  = (int)(idx / per);
    int c  = (int)(idx - (long long)e * per) << 2;
    int s  = src[e], d = dst[e];
    float nrm = dis[s] * ea[e] * dis[d];
    float4 m = *(const float4*)(M1 + (size_t)s * C + c);
    float* o = H + (size_t)d * C + c;
    atomicAdd(o + 0, m.x * nrm);
    atomicAdd(o + 1, m.y * nrm);
    atomicAdd(o + 2, m.z * nrm);
    atomicAdd(o + 3, m.w * nrm);
}

// Hh(f16) = relu(Hacc + dis[i]^2 * M1 + b1[c])  (fused self-loop+bias+relu,
// written directly in f16 so GEMM2/3 A-loads are pure b128s)
__global__ void relu_bias_self(const float* __restrict__ M1,
                               const float* __restrict__ Hacc,
                               const float* __restrict__ dis,
                               const float* __restrict__ b1,
                               _Float16* __restrict__ Hh, int N, int C) {
    long long i = (long long)blockIdx.x * blockDim.x + threadIdx.x;
    if (i >= (long long)N * C) return;
    int node = (int)(i / C);
    int c    = (int)(i - (long long)node * C);
    float ds = dis[node];
    float v  = Hacc[i] + ds * ds * M1[i] + b1[c];
    Hh[i] = (_Float16)(v > 0.f ? v : 0.f);
}

// ---------------------------------------------------------------------------
// Edge scatter for both heads:  mu[dst] += M2[src]*nrm ; lv[dst] += M3[src]*nrm
// ---------------------------------------------------------------------------
__global__ void scatter_lat(const float* __restrict__ M2,
                            const float* __restrict__ M3,
                            const int* __restrict__ src,
                            const int* __restrict__ dst,
                            const float* __restrict__ ea,
                            const float* __restrict__ dis,
                            float* __restrict__ mu, float* __restrict__ lv,
                            int E, int C) {
    const int per = C >> 2;
    long long idx = (long long)blockIdx.x * blockDim.x + threadIdx.x;
    if (idx >= (long long)E * per) return;
    int e  = (int)(idx / per);
    int c  = (int)(idx - (long long)e * per) << 2;
    int s  = src[e], d = dst[e];
    float nrm = dis[s] * ea[e] * dis[d];
    float4 m2 = *(const float4*)(M2 + (size_t)s * C + c);
    float4 m3 = *(const float4*)(M3 + (size_t)s * C + c);
    float* omu = mu + (size_t)d * C + c;
    float* olv = lv + (size_t)d * C + c;
    atomicAdd(omu + 0, m2.x * nrm); atomicAdd(omu + 1, m2.y * nrm);
    atomicAdd(omu + 2, m2.z * nrm); atomicAdd(omu + 3, m2.w * nrm);
    atomicAdd(olv + 0, m3.x * nrm); atomicAdd(olv + 1, m3.y * nrm);
    atomicAdd(olv + 2, m3.z * nrm); atomicAdd(olv + 3, m3.w * nrm);
}

// mu += dis^2*M2 + b_mu ; lv += dis^2*M3 + b_lv   (fused self-loop + bias)
__global__ void finalize_lat(const float* __restrict__ M2,
                             const float* __restrict__ M3,
                             const float* __restrict__ dis,
                             const float* __restrict__ bmu,
                             const float* __restrict__ blv,
                             float* __restrict__ mu, float* __restrict__ lv,
                             int N, int C) {
    long long i = (long long)blockIdx.x * blockDim.x + threadIdx.x;
    if (i >= (long long)N * C) return;
    int node = (int)(i / C);
    int c    = (int)(i - (long long)node * C);
    float d2 = dis[node] * dis[node];
    mu[i] += d2 * M2[i] + bmu[c];
    lv[i] += d2 * M3[i] + blv[c];
}

// ---------------------------------------------------------------------------
extern "C" void kernel_launch(void* const* d_in, const int* in_sizes, int n_in,
                              void* d_out, int out_size, void* d_ws, size_t ws_size,
                              hipStream_t stream) {
    const float* x   = (const float*)d_in[0];
    const int*   ei  = (const int*)  d_in[1];   // [2,E] (src row, dst row)
    const float* ea  = (const float*)d_in[2];
    const float* W1  = (const float*)d_in[3];
    const float* b1  = (const float*)d_in[4];
    const float* Wmu = (const float*)d_in[5];
    const float* bmu = (const float*)d_in[6];
    const float* Wlv = (const float*)d_in[7];
    const float* blv = (const float*)d_in[8];

    const int HID = in_sizes[4];               // 256
    const int LAT = in_sizes[6];               // 128
    const int IN  = in_sizes[3] / HID;         // 512
    const int N   = in_sizes[0] / IN;          // 50000
    const int E   = in_sizes[2];               // 800000

    const int* src = ei;
    const int* dst = ei + E;

    // workspace layout
    float* dis      = (float*)d_ws;                      // N f32
    float* M1       = dis + N;                           // N*HID f32 (reused: M2|M3)
    float* Hacc     = M1 + (size_t)N * HID;              // N*HID f32
    _Float16* Hh    = (_Float16*)(Hacc + (size_t)N * HID); // N*HID f16
    _Float16* W1t   = Hh + (size_t)N * HID;              // HID*IN  f16 (transposed)
    _Float16* Wmut  = W1t + (size_t)IN * HID;            // LAT*HID f16 (transposed)
    _Float16* Wlvt  = Wmut + (size_t)HID * LAT;          // LAT*HID f16 (transposed)

    float* M2 = M1;                            // N*LAT f32
    float* M3 = M1 + (size_t)N * LAT;          // N*LAT f32

    float* mu = (float*)d_out;                 // N*LAT
    float* lv = mu + (size_t)N * LAT;          // N*LAT

    const int BT = 256;
    auto blocks = [](long long n, int bt) { return (unsigned)((n + bt - 1) / bt); };

    // 1) init: deg=1.0 (self loop), Hacc=0, out=0; transpose weights to f16
    fill_kernel<<<blocks(N, BT), BT, 0, stream>>>(dis, 1.0f, N);
    fill_kernel<<<blocks((long long)N * HID, BT), BT, 0, stream>>>(Hacc, 0.0f, (long long)N * HID);
    fill_kernel<<<blocks((long long)N * LAT * 2, BT), BT, 0, stream>>>(mu, 0.0f, (long long)N * LAT * 2);
    transpose_f16_kernel<<<blocks((long long)IN * HID, BT), BT, 0, stream>>>(W1, W1t, IN, HID);
    transpose_f16_kernel<<<blocks((long long)HID * LAT, BT), BT, 0, stream>>>(Wmu, Wmut, HID, LAT);
    transpose_f16_kernel<<<blocks((long long)HID * LAT, BT), BT, 0, stream>>>(Wlv, Wlvt, HID, LAT);

    // 2) degree + symmetric norm factors
    degree_kernel<<<blocks(E, BT), BT, 0, stream>>>(dst, ea, dis, E);
    invsqrt_kernel<<<blocks(N, BT), BT, 0, stream>>>(dis, N);

    // 3) M1 = X @ W1   (WMMA, 16x64 strip per wave)
    {
        long long waves = (long long)(N / 16) * (HID / 64);
        gemm_wmma_f16<float><<<blocks(waves * 32, BT), BT, 0, stream>>>(x, W1t, M1, N, IN, HID);
    }

    // 4) edge aggregation into Hacc, then fused self-loop + bias + relu -> f16
    scatter_hid<<<blocks((long long)E * (HID / 4), BT), BT, 0, stream>>>(M1, src, dst, ea, dis, Hacc, E, HID);
    relu_bias_self<<<blocks((long long)N * HID, BT), BT, 0, stream>>>(M1, Hacc, dis, b1, Hh, N, HID);

    // 5) M2 = H @ W_mu ; M3 = H @ W_lv   (WMMA, f16 A operand)
    {
        long long waves = (long long)(N / 16) * (LAT / 64);
        gemm_wmma_f16<_Float16><<<blocks(waves * 32, BT), BT, 0, stream>>>(Hh, Wmut, M2, N, HID, LAT);
        gemm_wmma_f16<_Float16><<<blocks(waves * 32, BT), BT, 0, stream>>>(Hh, Wlvt, M3, N, HID, LAT);
    }

    // 6) edge aggregation into mu/lv + fused self-loop + bias
    scatter_lat<<<blocks((long long)E * (LAT / 4), BT), BT, 0, stream>>>(M2, M3, src, dst, ea, dis, mu, lv, E, LAT);
    finalize_lat<<<blocks((long long)N * LAT, BT), BT, 0, stream>>>(M2, M3, dis, bmu, blv, mu, lv, N, LAT);
}
